// TestScaledMMRSModel_71717363909052
// MI455X (gfx1250) — 8-bit matrix operand order repaired and checked by reading, compiled, not run on hardware
//
#include <hip/hip_runtime.h>
#include <hip/hip_bf16.h>
#include <math.h>


typedef __attribute__((ext_vector_type(16))) int   v16i;
typedef __attribute__((ext_vector_type(8)))  float v8f;

#if __has_builtin(__builtin_amdgcn_global_load_async_to_lds_b128)
#define HAVE_ASYNC_LDS 1
typedef int vec4i __attribute__((vector_size(16)));
typedef __attribute__((address_space(1))) vec4i g_vec4i;
typedef __attribute__((address_space(3))) vec4i l_vec4i;
#endif

template <int N>
__device__ __forceinline__ void wait_async() {
#if __has_builtin(__builtin_amdgcn_s_wait_asynccnt)
  __builtin_amdgcn_s_wait_asynccnt(N);
#elif defined(HAVE_ASYNC_LDS)
  asm volatile("s_wait_asynccnt %0" ::"i"(N) : "memory");
#endif
}


__device__ inline unsigned int f32_to_e4m3_sw(float x) {
  unsigned int u = __float_as_uint(x);
  unsigned int s = (u >> 24) & 0x80u;
  float ax = __uint_as_float(u & 0x7fffffffu);
  if (!(ax == ax)) return s | 0x7fu;
  if (ax >= 448.0f) return s | 0x7eu;
  int E = (int)((u >> 23) & 0xffu) - 127;
  if (E < -6) {
    unsigned int q = (unsigned int)rintf(ax * 512.0f);
    return s | (q & 0x7fu);
  }
  float q = rintf(ax * exp2f((float)(3 - E)));
  unsigned int Ef = (unsigned int)(E + 7);
  unsigned int mant;
  if (q >= 16.0f) { Ef += 1u; mant = 0u; } else { mant = (unsigned int)q - 8u; }
  if (Ef > 15u) return s | 0x7eu;
  return s | (Ef << 3) | mant;
}

__global__ __launch_bounds__(256) void quantize_fp8_kernel(
    const float* __restrict__ src, unsigned int* __restrict__ dst, size_t n4) {
  size_t i = (size_t)blockIdx.x * blockDim.x + threadIdx.x;
  if (i >= n4) return;
  float4 f = reinterpret_cast<const float4*>(src)[i];
  unsigned int w;
#if __has_builtin(__builtin_amdgcn_cvt_pk_fp8_f32)
  w = 0;
  w = __builtin_amdgcn_cvt_pk_fp8_f32(f.x, f.y, w, false);
  w = __builtin_amdgcn_cvt_pk_fp8_f32(f.z, f.w, w, true);
#else
  w = f32_to_e4m3_sw(f.x) | (f32_to_e4m3_sw(f.y) << 8) |
      (f32_to_e4m3_sw(f.z) << 16) | (f32_to_e4m3_sw(f.w) << 24);
#endif
  dst[i] = w;
}


#define BM 128
#define BN 128
#define KS 128
#define LDA 144
#define LDB 144
#define NBUF 3

__global__ __launch_bounds__(256)
__attribute__((amdgpu_waves_per_eu(2, 8)))
void gemm_fp8_rs_kernel(
    const unsigned char* __restrict__ Aq,
    const unsigned char* __restrict__ Bq,
    const float* __restrict__ scale_b,
    _Float16* __restrict__ out,
    int Mout, int N, int K, int R) {
  __shared__ unsigned char sA[NBUF][BM * LDA];
  __shared__ unsigned char sB[NBUF][KS * LDB];

  const int tid  = threadIdx.x;
  const int lane = tid & 31;
  const int wave = tid >> 5;
  const int wm   = wave & 3;
  const int wn   = wave >> 2;
  const int lm   = lane & 15;
  const int lhi  = lane >> 4;

  const int bn0 = blockIdx.x * BN;
  const int bm0 = blockIdx.y * BM;

  v8f acc[2][4] = {};

  const int ksteps = K / KS;
  const int nsteps = R * ksteps;

  auto advance = [&](int& r, int& kk) {
    kk += KS;
    if (kk == K) { kk = 0; ++r; }
  };

  auto compute = [&](int buf) {
    union { v16i v; unsigned long long q[8]; } af[2];
#pragma unroll
    for (int mt = 0; mt < 2; ++mt) {
      const unsigned char* base = &sA[buf][(wm * 32 + mt * 16 + lm) * LDA];
#pragma unroll
      for (int c = 0; c < 8; ++c) {
        const int kb = ((c >> 2) << 6) + ((c & 3) << 4) + (lhi << 3);
        af[mt].q[c] = *reinterpret_cast<const unsigned long long*>(base + kb);
      }
    }
#pragma unroll
    for (int nt = 0; nt < 4; ++nt) {
      union { v16i v; unsigned char b[64]; } bf;
#pragma unroll
      for (int s = 0; s < 64; ++s)
        bf.b[s] = sB[buf][(((s >> 5) << 6) + (((s >> 3) & 3) << 4) + (lhi << 3) + (s & 7)) * LDB
            + wn * 64 + nt * 16 + lm];
#pragma unroll
      for (int mt = 0; mt < 2; ++mt)
        acc[mt][nt] = __builtin_amdgcn_wmma_f32_16x16x128_fp8_fp8(
            af[mt].v, bf.v, (short)0, acc[mt][nt], false, false);
    }
  };

#ifdef HAVE_ASYNC_LDS
  auto issue_async = [&](int buf, int r, int kk) {
#pragma unroll
    for (int j = 0; j < 4; ++j) {
      const int seg = tid + 256 * j;
      const int row = seg >> 3;
      const int col = (seg & 7) << 4;
      __builtin_amdgcn_global_load_async_to_lds_b128(
          (g_vec4i*)(Aq + (size_t)(r * Mout + bm0 + row) * K + kk + col),
          (l_vec4i*)(&sA[buf][row * LDA + col]), 0, 0);
      __builtin_amdgcn_global_load_async_to_lds_b128(
          (g_vec4i*)(Bq + (size_t)(kk + row) * N + bn0 + col),
          (l_vec4i*)(&sB[buf][row * LDB + col]), 0, 0);
    }
  };
  auto prefetch_step = [&](int r, int kk) {
    __builtin_prefetch(
        Aq + (size_t)(r * Mout + bm0 + (tid >> 1)) * K + kk + (tid & 1) * 64, 0, 1);
    __builtin_prefetch(
        Bq + (size_t)(kk + (tid >> 1)) * N + bn0 + (tid & 1) * 64, 0, 1);
  };

  int r_i = 0, kk_i = 0;
  int r_p = 0, kk_p = 0;
  for (int t = 0; t < 4 && t < nsteps; ++t) advance(r_p, kk_p);

  issue_async(0, r_i, kk_i); advance(r_i, kk_i);
  if (nsteps > 1) { issue_async(1, r_i, kk_i); advance(r_i, kk_i); }
  wait_async<8>();
  __syncthreads();

  int cb = 0, ib = 2 % NBUF;
  for (int s = 0; s < nsteps; ++s) {
    const bool more = (s + 2 < nsteps);
    if (more) {
      issue_async(ib, r_i, kk_i);
      advance(r_i, kk_i);
      if (s + 4 < nsteps) { prefetch_step(r_p, kk_p); advance(r_p, kk_p); }
    }

    compute(cb);

    if (s + 1 < nsteps) {
      if (more) wait_async<8>();
      else      wait_async<0>();
      __syncthreads();
    }
    if (++cb == NBUF) cb = 0;
    if (++ib == NBUF) ib = 0;
  }
#else
  uint4 ra[4], rb[4];
  int r_l = 0, kk_l = 0;
  auto load_tiles = [&]() {
#pragma unroll
    for (int j = 0; j < 4; ++j) {
      const int seg = tid + 256 * j;
      const int row = seg >> 3;
      const int col = (seg & 7) << 4;
      ra[j] = *reinterpret_cast<const uint4*>(
          Aq + (size_t)(r_l * Mout + bm0 + row) * K + kk_l + col);
      rb[j] = *reinterpret_cast<const uint4*>(
          Bq + (size_t)(kk_l + row) * N + bn0 + col);
    }
    advance(r_l, kk_l);
  };
  auto store_tiles = [&](int buf) {
#pragma unroll
    for (int j = 0; j < 4; ++j) {
      const int seg = tid + 256 * j;
      const int row = seg >> 3;
      const int col = (seg & 7) << 4;
      *reinterpret_cast<uint4*>(&sA[buf][row * LDA + col]) = ra[j];
      *reinterpret_cast<uint4*>(&sB[buf][row * LDB + col]) = rb[j];
    }
  };

  load_tiles();
  store_tiles(0);
  __syncthreads();

  for (int s = 0; s < nsteps; ++s) {
    if (s + 1 < nsteps) load_tiles();
    compute(s % NBUF);
    __syncthreads();
    if (s + 1 < nsteps) {
      store_tiles((s + 1) % NBUF);
      __syncthreads();
    }
  }
#endif

#pragma unroll
  for (int nt = 0; nt < 4; ++nt) {
    const int n = bn0 + wn * 64 + nt * 16 + lm;
    const float sc = scale_b[n];
#pragma unroll
    for (int mt = 0; mt < 2; ++mt) {
#pragma unroll
      for (int v = 0; v < 8; ++v) {
        const int m = bm0 + wm * 32 + mt * 16 + v + lhi * 8;
        out[(size_t)m * N + n] = (_Float16)(acc[mt][nt][v] * sc);
      }
    }
  }
}


extern "C" void kernel_launch(void* const* d_in, const int* in_sizes, int n_in,
                              void* d_out, int out_size, void* d_ws, size_t ws_size,
                              hipStream_t stream) {
  const float* input   = (const float*)d_in[0];
  const float* weight  = (const float*)d_in[1];
  const float* scale_b = (const float*)d_in[2];

  int H = 1;
  while ((long long)H * H < (long long)in_sizes[1]) H <<= 1;
  const int Mtot = in_sizes[0] / H;
  const int R = 8;
  const int Mout = Mtot / R;

  unsigned char* Aq = (unsigned char*)d_ws;
  unsigned char* Bq = Aq + (size_t)Mtot * H;

  const size_t n4a = (size_t)Mtot * H / 4;
  const size_t n4b = (size_t)H * H / 4;
  quantize_fp8_kernel<<<(unsigned)((n4a + 255) / 256), 256, 0, stream>>>(
      input, (unsigned int*)Aq, n4a);
  quantize_fp8_kernel<<<(unsigned)((n4b + 255) / 256), 256, 0, stream>>>(
      weight, (unsigned int*)Bq, n4b);

  dim3 grid(H / BN, Mout / BM);
  gemm_fp8_rs_kernel<<<grid, 256, 0, stream>>>(
      Aq, Bq, scale_b, (_Float16*)d_out, Mout, H, H, R);
}
